// VQVAE_42056319762856
// MI455X (gfx1250) — compile-verified, matrix operand order checked by reading, not run on hardware
//
#include <hip/hip_runtime.h>
#include <hip/hip_bf16.h>

// ---------------------------------------------------------------------------
// VQ-VAE forward for MI455X (gfx1250): bf16 WMMA GEMMs with async LDS-DMA
// staging (GLOBAL_LOAD_ASYNC_TO_LDS_B128, ASYNCcnt) + fused VQ argmin.
// N=32768, D=H=K=1024.
// ---------------------------------------------------------------------------

typedef __bf16 bf16_t;
typedef __attribute__((ext_vector_type(16))) __bf16 v16bf;
typedef __attribute__((ext_vector_type(8)))  __bf16 v8bf;
typedef __attribute__((ext_vector_type(8)))  float  v8f;

__device__ __forceinline__ v16bf make16(v8bf lo, v8bf hi) {
    v16bf r;
#pragma unroll
    for (int i = 0; i < 8; ++i) { r[i] = lo[i]; r[i + 8] = hi[i]; }
    return r;
}

// Generic LDS pointer -> hardware LDS byte offset (aperture: addr[31:0]).
__device__ __forceinline__ unsigned lds_off(const void* p) {
    return (unsigned)(unsigned long long)p;
}

// Async global -> LDS 16B copy, tracked by ASYNCcnt (no VGPR data path).
__device__ __forceinline__ void async_b128(unsigned ldsByteOff, const void* gaddr) {
    asm volatile("global_load_async_to_lds_b128 %0, %1, off"
                 :: "v"(ldsByteOff), "v"(gaddr) : "memory");
}

__device__ __forceinline__ void wait_async0() {
    asm volatile("s_wait_asynccnt 0x0" ::: "memory");
}

// ---------------------------------------------------------------------------
// f32 -> bf16 convert, 8 elements/thread/iter (two float4 in, one b128 out)
// ---------------------------------------------------------------------------
__global__ void k_convert(const float4* __restrict__ in, v8bf* __restrict__ out,
                          long count8) {
    long i = (long)blockIdx.x * blockDim.x + threadIdx.x;
    long stride = (long)gridDim.x * blockDim.x;
    for (; i < count8; i += stride) {
        float4 a = in[2 * i], b = in[2 * i + 1];
        v8bf o;
        o[0] = (bf16_t)a.x; o[1] = (bf16_t)a.y; o[2] = (bf16_t)a.z; o[3] = (bf16_t)a.w;
        o[4] = (bf16_t)b.x; o[5] = (bf16_t)b.y; o[6] = (bf16_t)b.z; o[7] = (bf16_t)b.w;
        out[i] = o;
    }
}

// ---------------------------------------------------------------------------
// Transpose + convert: Wt[n*Kd + k] = (bf16) W[k*Nd + n]  (32x32 LDS tiles)
// ---------------------------------------------------------------------------
__global__ void k_transpose(const float* __restrict__ W, bf16_t* __restrict__ Wt,
                            int Kd, int Nd) {
    __shared__ float tile[32][33];
    const int n0 = blockIdx.x * 32;
    const int k0 = blockIdx.y * 32;
    const int x = threadIdx.x, y = threadIdx.y;
#pragma unroll
    for (int r = 0; r < 32; r += 8)
        tile[y + r][x] = W[(size_t)(k0 + y + r) * Nd + n0 + x];
    __syncthreads();
#pragma unroll
    for (int r = 0; r < 32; r += 8)
        Wt[(size_t)(n0 + y + r) * Kd + k0 + x] = (bf16_t)tile[x][y + r];
}

// ---------------------------------------------------------------------------
// Codebook per-row sum and sum-of-squares.
// ---------------------------------------------------------------------------
__global__ void k_rowstats(const float* __restrict__ C, float* __restrict__ rs,
                           float* __restrict__ rq, int Hd) {
    __shared__ float s1[256], s2[256];
    const int row = blockIdx.x;
    const int t = threadIdx.x;
    float a = 0.f, b = 0.f;
    for (int j = t; j < Hd; j += 256) {
        float v = C[(size_t)row * Hd + j];
        a += v; b += v * v;
    }
    s1[t] = a; s2[t] = b;
    __syncthreads();
    for (int s = 128; s > 0; s >>= 1) {
        if (t < s) { s1[t] += s1[t + s]; s2[t] += s2[t + s]; }
        __syncthreads();
    }
    if (t == 0) { rs[row] = s1[0]; rq[row] = s2[0]; }
}

// ---------------------------------------------------------------------------
// Fused VQ: dist[i] = Hd*z^2 - 2 z rs[i] + rq[i]; argmin (first-occurrence
// tie-break) then gather codebook row -> f32 output + bf16 for next GEMM.
// ---------------------------------------------------------------------------
__global__ void k_vq(const float* __restrict__ z, const float* __restrict__ C,
                     const float* __restrict__ rs, const float* __restrict__ rq,
                     float* __restrict__ zq, bf16_t* __restrict__ zqb,
                     int Kn, int Hd) {
    __shared__ float sd[256];
    __shared__ int   si[256];
    const int n = blockIdx.x;
    const int t = threadIdx.x;
    const float* zr = z + (size_t)n * Kn;
    const float hF = (float)Hd;
    float bd = 3.402823466e38f;
    int   bi = 0x7fffffff;
    for (int i = t; i < Kn; i += 256) {
        float zv = zr[i];
        float d = hF * zv * zv - 2.f * zv * rs[i] + rq[i];
        if (d < bd) { bd = d; bi = i; }
    }
    sd[t] = bd; si[t] = bi;
    __syncthreads();
    for (int s = 128; s > 0; s >>= 1) {
        if (t < s) {
            float d2 = sd[t + s]; int i2 = si[t + s];
            if (d2 < sd[t] || (d2 == sd[t] && i2 < si[t])) { sd[t] = d2; si[t] = i2; }
        }
        __syncthreads();
    }
    const int best = si[0];
    const float* cr = C + (size_t)best * Hd;
    float*  zo = zq  + (size_t)n * Hd;
    bf16_t* zb = zqb + (size_t)n * Hd;
    for (int j = t; j < Hd; j += 256) {
        float v = cr[j];
        zo[j] = v;
        zb[j] = (bf16_t)v;
    }
}

// ---------------------------------------------------------------------------
// bf16 WMMA GEMM:  out = act(A[M,Kd] * Bt[Nd,Kd]^T + bias)
//   Block tile 128x128, K-step 32, 8 wave32 waves, each wave 32x64 = 2x4
//   WMMA tiles. Double-buffered LDS; tiles staged with async LDS-DMA
//   (global_load_async_to_lds_b128) overlapping the WMMAs of the live tile.
//   ACT: 0=relu, 1=sigmoid.
// ---------------------------------------------------------------------------
template <int ACT, bool WF32, bool WBF16>
__global__ __launch_bounds__(256) void k_gemm(const bf16_t* __restrict__ A,
                                              const bf16_t* __restrict__ Bt,
                                              const float* __restrict__ bias,
                                              float* __restrict__ outF,
                                              bf16_t* __restrict__ outB,
                                              int M, int Nd, int Kd) {
    constexpr int LDSK  = 40;                         // 32 + 8 pad (bank spread)
    constexpr int TILE  = 128 * LDSK;                 // elements per buffer
    constexpr unsigned BUFB = TILE * sizeof(bf16_t);  // bytes per buffer
    __shared__ __attribute__((aligned(16))) bf16_t sA[2 * TILE];
    __shared__ __attribute__((aligned(16))) bf16_t sB[2 * TILE];

    const int t     = threadIdx.x;
    const int lane  = t & 31;
    const int wave  = t >> 5;
    const int waveM = wave >> 1;   // 0..3  (32-row strip)
    const int waveN = wave & 1;    // 0..1  (64-col strip)
    const int m0 = blockIdx.y * 128;
    const int n0 = blockIdx.x * 128;
    const int lrow  = lane & 15;
    const int lhalf = lane >> 4;
    const int kb    = lhalf << 3;  // WMMA 16-bit operand K base (0 or 8)

    // ---- per-thread staging assignment: 2 fixed 16B chunks per matrix ----
    const int cRow = t >> 2;            // 0..63
    const int cCol = (t & 3) << 3;      // 0,8,16,24 elements
    const bf16_t* gA0 = A  + (size_t)(m0 + cRow) * Kd + cCol;
    const bf16_t* gA1 = gA0 + (size_t)64 * Kd;
    const bf16_t* gB0 = Bt + (size_t)(n0 + cRow) * Kd + cCol;
    const bf16_t* gB1 = gB0 + (size_t)64 * Kd;
    const unsigned lA0 = lds_off(&sA[cRow * LDSK + cCol]);
    const unsigned lA1 = lds_off(&sA[(cRow + 64) * LDSK + cCol]);
    const unsigned lB0 = lds_off(&sB[cRow * LDSK + cCol]);
    const unsigned lB1 = lds_off(&sB[(cRow + 64) * LDSK + cCol]);

    v8f acc[2][4] = {};

    // ---- prologue: async-stage tile 0 into buffer 0 ----
    async_b128(lA0, gA0); async_b128(lA1, gA1);
    async_b128(lB0, gB0); async_b128(lB1, gB1);

    auto step = [&](int p, bool stageNext) {
        wait_async0();      // this wave's DMA into buffer p has landed
        __syncthreads();    // everyone's has
        if (stageNext) {    // DMA tile k+1 into the other buffer, under WMMAs
            gA0 += 32; gA1 += 32; gB0 += 32; gB1 += 32;
            const unsigned q = (unsigned)(p ^ 1) * BUFB;
            async_b128(lA0 + q, gA0); async_b128(lA1 + q, gA1);
            async_b128(lB0 + q, gB0); async_b128(lB1 + q, gB1);
        }
        const bf16_t* cA = sA + p * TILE;
        const bf16_t* cB = sB + p * TILE;
        v16bf afrag[2], bfrag[4];
#pragma unroll
        for (int mi = 0; mi < 2; ++mi) {
            const int row = waveM * 32 + mi * 16 + lrow;
            afrag[mi] = make16(*(const v8bf*)&cA[row * LDSK + kb],
                               *(const v8bf*)&cA[row * LDSK + kb + 16]);
        }
#pragma unroll
        for (int ni = 0; ni < 4; ++ni) {
            const int col = waveN * 64 + ni * 16 + lrow;
            bfrag[ni] = make16(*(const v8bf*)&cB[col * LDSK + kb],
                               *(const v8bf*)&cB[col * LDSK + kb + 16]);
        }
#pragma unroll
        for (int mi = 0; mi < 2; ++mi)
#pragma unroll
            for (int ni = 0; ni < 4; ++ni)
                acc[mi][ni] = __builtin_amdgcn_wmma_f32_16x16x32_bf16(
                    false, afrag[mi], false, bfrag[ni],
                    (short)0, acc[mi][ni], false, false);
    };

    const int tiles = Kd >> 5;              // K multiple of 64 -> even
    for (int j = 0; j < tiles; j += 2) {
        step(0, true);
        step(1, j + 2 < tiles);
    }

    // ---- epilogue: bias + activation, store f32 and/or bf16 ----
#pragma unroll
    for (int mi = 0; mi < 2; ++mi) {
#pragma unroll
        for (int ni = 0; ni < 4; ++ni) {
            const int gn = n0 + waveN * 64 + ni * 16 + lrow;
            const float bn = bias[gn];
            const size_t base =
                (size_t)(m0 + waveM * 32 + mi * 16 + lhalf * 8) * Nd + gn;
#pragma unroll
            for (int r = 0; r < 8; ++r) {
                float v = acc[mi][ni][r] + bn;
                if (ACT == 0) v = v > 0.f ? v : 0.f;
                else          v = 1.f / (1.f + __expf(-v));
                if constexpr (WF32)  outF[base + (size_t)r * Nd] = v;
                if constexpr (WBF16) outB[base + (size_t)r * Nd] = (bf16_t)v;
            }
        }
    }
}

// ---------------------------------------------------------------------------
// Orchestration
// ---------------------------------------------------------------------------
extern "C" void kernel_launch(void* const* d_in, const int* in_sizes, int n_in,
                              void* d_out, int out_size, void* d_ws, size_t ws_size,
                              hipStream_t stream) {
    const float* x  = (const float*)d_in[0];
    const float* W1 = (const float*)d_in[1];
    const float* b1 = (const float*)d_in[2];
    const float* W2 = (const float*)d_in[3];
    const float* b2 = (const float*)d_in[4];
    const float* cb = (const float*)d_in[5];
    const float* W3 = (const float*)d_in[6];
    const float* b3 = (const float*)d_in[7];
    const float* W4 = (const float*)d_in[8];
    const float* b4 = (const float*)d_in[9];

    const int Nn = 32768, Dd = 1024, Hh = 1024, Kk = 1024;

    float* out_recon = (float*)d_out;                     // (N, D)
    float* out_ze    = out_recon + (size_t)Nn * Dd;       // (N, H)
    float* out_zq    = out_ze    + (size_t)Nn * Hh;       // (N, H)

    char* ws = (char*)d_ws;
    bf16_t* Wt1 = (bf16_t*)(ws + (0ull  << 20));
    bf16_t* Wt2 = (bf16_t*)(ws + (2ull  << 20));
    bf16_t* Wt3 = (bf16_t*)(ws + (4ull  << 20));
    bf16_t* Wt4 = (bf16_t*)(ws + (6ull  << 20));
    float*  rs  = (float*) (ws + (8ull  << 20));
    float*  rq  = rs + Kk;
    bf16_t* bufA = (bf16_t*)(ws + (16ull << 20));          // xb, then z_q bf16
    bf16_t* bufB = bufA + (size_t)Nn * Hh;                 // h1 bf16, then h3 bf16

    const dim3 tb(32, 8);
    k_transpose<<<dim3(Hh / 32, Dd / 32), tb, 0, stream>>>(W1, Wt1, Dd, Hh);
    k_transpose<<<dim3(Hh / 32, Hh / 32), tb, 0, stream>>>(W2, Wt2, Hh, Hh);
    k_transpose<<<dim3(Hh / 32, Hh / 32), tb, 0, stream>>>(W3, Wt3, Hh, Hh);
    k_transpose<<<dim3(Dd / 32, Hh / 32), tb, 0, stream>>>(W4, Wt4, Hh, Dd);

    k_convert<<<2048, 256, 0, stream>>>((const float4*)x, (v8bf*)bufA,
                                        (long)Nn * Dd / 8);

    // h1 = relu(x @ W1 + b1) -> bf16
    k_gemm<0, false, true><<<dim3(Hh / 128, Nn / 128), 256, 0, stream>>>(
        bufA, Wt1, b1, (float*)nullptr, bufB, Nn, Hh, Dd);
    // z_e = relu(h1 @ W2 + b2) -> f32 output
    k_gemm<0, true, false><<<dim3(Hh / 128, Nn / 128), 256, 0, stream>>>(
        bufB, Wt2, b2, out_ze, (bf16_t*)nullptr, Nn, Hh, Hh);

    // VQ: codebook row stats, then fused argmin + gather
    k_rowstats<<<Kk, 256, 0, stream>>>(cb, rs, rq, Hh);
    k_vq<<<Nn, 256, 0, stream>>>(out_ze, cb, rs, rq, out_zq, bufA, Kk, Hh);

    // h3 = relu(z_q @ W3 + b3) -> bf16
    k_gemm<0, false, true><<<dim3(Hh / 128, Nn / 128), 256, 0, stream>>>(
        bufA, Wt3, b3, (float*)nullptr, bufB, Nn, Hh, Hh);
    // x_recon = sigmoid(h3 @ W4 + b4) -> f32 output
    k_gemm<1, true, false><<<dim3(Dd / 128, Nn / 128), 256, 0, stream>>>(
        bufB, Wt4, b4, out_recon, (bf16_t*)nullptr, Nn, Dd, Hh);
}
